// Multi_Head_Attention_Layer_36636071034945
// MI455X (gfx1250) — matrix operand order repaired and checked by reading, not compiled, not run on hardware
//
#include <hip/hip_runtime.h>
#include <stdint.h>

// Problem constants
#define B_  4
#define S_  2048
#define D_  1024
#define H_  16
#define DH_ 64

typedef __attribute__((ext_vector_type(16))) __bf16 v16bf;
typedef __attribute__((ext_vector_type(8)))  float  v8f;

// Async global->LDS path (gfx1250): guarded so absence degrades to plain copies
#if defined(__has_builtin)
#if __has_builtin(__builtin_amdgcn_global_load_async_to_lds_b128) && \
    __has_builtin(__builtin_amdgcn_s_wait_asynccnt)
#define USE_ASYNC_LDS 1
#endif
#endif
#ifndef USE_ASYNC_LDS
#define USE_ASYNC_LDS 0
#endif

#if USE_ASYNC_LDS
// Builtin signature (from clang diagnostic): params are pointers to
// int __attribute__((vector_size(16))) in addrspace(1) (global) / (3) (LDS).
typedef int v4i_ __attribute__((vector_size(16)));
typedef __attribute__((address_space(1))) v4i_ gv4i;
typedef __attribute__((address_space(3))) v4i_ lv4i;
#endif

union BF16Frag {
  v16bf v;
  unsigned int u[8];
  uint4 q[2];
};

// round-to-nearest-even fp32 -> bf16, bit-exact, no __bf16 arithmetic needed
__device__ __forceinline__ unsigned short f2bf(float f) {
  union { float f; unsigned int u; } c; c.f = f;
  unsigned int u = c.u;
  unsigned int r = (u + 0x7FFFu + ((u >> 16) & 1u)) >> 16;
  return (unsigned short)r;
}

// 16-bit A-matrix (16x32) k order used here (not AMD's table order): lane half (lanes>=16) shifts K by 16,
// VGPRs 0..7 cover K 0..15 (pairs) within the half.
__device__ __forceinline__ int a_kbase(int v, int hl) {
  return (hl << 4) + (v << 1);
}

__device__ __forceinline__ float redmax16(float x) {
#pragma unroll
  for (int m = 1; m < 16; m <<= 1) x = fmaxf(x, __shfl_xor(x, m, 32));
  return x;
}
__device__ __forceinline__ float redsum16(float x) {
#pragma unroll
  for (int m = 1; m < 16; m <<= 1) x += __shfl_xor(x, m, 32);
  return x;
}

// ---------------------------------------------------------------------------
// fp32 -> bf16 conversion (streaming: NT loads, data is read exactly once)
// ---------------------------------------------------------------------------
__global__ void cvt_f32_bf16(const float* __restrict__ in,
                             unsigned short* __restrict__ out, int n) {
  int i = blockIdx.x * blockDim.x + threadIdx.x;
  if (i < n) out[i] = f2bf(__builtin_nontemporal_load(&in[i]));
}

// ---------------------------------------------------------------------------
// Per-head transpose: Vt[b][h][dh][s] = V[b][s][h*DH+dh]
// ---------------------------------------------------------------------------
__global__ void transpose_v(const unsigned short* __restrict__ Vin,
                            unsigned short* __restrict__ Vt) {
  int i = blockIdx.x * 256 + threadIdx.x;   // linear over B*H*DH*S (exact)
  int s  = i % S_;
  int r  = i / S_;
  int dh = r % DH_;
  int bh = r / DH_;
  int b = bh / H_, h = bh % H_;
  Vt[i] = Vin[((size_t)b * S_ + s) * D_ + h * DH_ + dh];
}

// ---------------------------------------------------------------------------
// bf16 GEMM: C = A(MxK) * B(KxN) + bias, A/B bf16, accumulate f32.
// Block: 256 threads (8 waves). Block tile 128(M) x 64(N), K-step 32.
// Each wave: one A fragment (16x32) x four B fragments (32x16) -> 4 WMMAs.
// A tile staged with GLOBAL_LOAD_ASYNC_TO_LDS_B128 when available.
// ---------------------------------------------------------------------------
template <bool BF16OUT>
__global__ __launch_bounds__(256) void gemm_bf16(
    const unsigned short* __restrict__ A,
    const unsigned short* __restrict__ Bm,
    const float* __restrict__ bias,
    float* __restrict__ Cf,
    unsigned short* __restrict__ Cbf,
    int M, int N, int K) {
  __shared__ unsigned short As[128 * 32];   // [m][k] row-major
  __shared__ unsigned short Bs[64][34];     // [n][k] transposed, padded

  const int t = threadIdx.x;
  const int wv = t >> 5, ln = t & 31;
  const int hl = ln >> 4, mrow = ln & 15, ncol = ln & 15;
  const int m0 = blockIdx.x * 128, n0 = blockIdx.y * 64;
  const int wrow = wv * 16 + mrow;

  const int arow = t >> 2, acol = (t & 3) * 8;  // A tile: 64 rows/pass, 2 passes
  const int brow = t >> 3, bcol = (t & 7) * 8;  // B tile: 32 rows, 1 pass

  v8f c[4] = {v8f{}, v8f{}, v8f{}, v8f{}};

  for (int k0 = 0; k0 < K; k0 += 32) {
    __syncthreads();
#if USE_ASYNC_LDS
    {
      __builtin_amdgcn_global_load_async_to_lds_b128(
          (gv4i*)(uintptr_t)&A[(size_t)(m0 + arow) * K + k0 + acol],
          (lv4i*)(uintptr_t)&As[arow * 32 + acol], 0, 0);
      __builtin_amdgcn_global_load_async_to_lds_b128(
          (gv4i*)(uintptr_t)&A[(size_t)(m0 + arow + 64) * K + k0 + acol],
          (lv4i*)(uintptr_t)&As[(arow + 64) * 32 + acol], 0, 0);
    }
#else
    // Stage A (128x32) row-major via b128 copies
    *(uint4*)&As[arow * 32 + acol] =
        *(const uint4*)&A[(size_t)(m0 + arow) * K + k0 + acol];
    *(uint4*)&As[(arow + 64) * 32 + acol] =
        *(const uint4*)&A[(size_t)(m0 + arow + 64) * K + k0 + acol];
#endif
    // Stage B (32x64) transposed into [n][k]
    {
      uint4 w = *(const uint4*)&Bm[(size_t)(k0 + brow) * N + n0 + bcol];
      const unsigned short* wp = (const unsigned short*)&w;
#pragma unroll
      for (int i = 0; i < 8; ++i) Bs[bcol + i][brow] = wp[i];
    }
    if (k0 + 32 < K) {  // global_prefetch_b8 of next tiles
      __builtin_prefetch(&A[(size_t)(m0 + arow) * K + k0 + 32 + acol], 0, 3);
      __builtin_prefetch(&Bm[(size_t)(k0 + 32 + brow) * N + n0 + bcol], 0, 3);
    }
#if USE_ASYNC_LDS
    __builtin_amdgcn_s_wait_asynccnt(0);
#endif
    __syncthreads();

    // A fragment: lane M = mrow (both halves), K pairs in contiguous order (a_kbase)
    BF16Frag a;
#pragma unroll
    for (int v = 0; v < 8; ++v)
      a.u[v] = *(const unsigned int*)&As[wrow * 32 + a_kbase(v, hl)];

    // B fragments: lane = N column, half selects K 0..15 / 16..31
    BF16Frag bf[4];
#pragma unroll
    for (int j = 0; j < 4; ++j) {
      int n = j * 16 + ncol;
#pragma unroll
      for (int v = 0; v < 8; ++v)
        bf[j].u[v] = *(const unsigned int*)&Bs[n][(hl << 4) + (v << 1)];
    }

#pragma unroll
    for (int j = 0; j < 4; ++j)
      c[j] = __builtin_amdgcn_wmma_f32_16x16x32_bf16(
          false, a.v, false, bf[j].v, (short)0, c[j], false, false);
  }

  // Epilogue: C layout = VGPR r -> row (r or r+8), lane%16 -> col
#pragma unroll
  for (int r = 0; r < 8; ++r) {
    int gm = m0 + wv * 16 + r + hl * 8;
#pragma unroll
    for (int j = 0; j < 4; ++j) {
      int gn = n0 + j * 16 + ncol;
      float val = c[j][r] + bias[gn];
      if (BF16OUT) Cbf[(size_t)gm * N + gn] = f2bf(val);
      else         __builtin_nontemporal_store(val, &Cf[(size_t)gm * N + gn]);
    }
  }
}

// ---------------------------------------------------------------------------
// Flash attention: block = 4 waves (128 thr); each wave owns 16 query rows.
// Per 32-key chunk: 4 score WMMAs (K-dim = DH=64 split in 2), online softmax,
// P staged through LDS (C-layout -> A-layout), 4 PV WMMAs.
// Q [B,S,D] bf16, K [B,S,D] bf16 (natural), Vt [B,H,DH,S] bf16, O [B,S,D] bf16
// ---------------------------------------------------------------------------
__global__ __launch_bounds__(128) void flash_attn(
    const unsigned short* __restrict__ Q,
    const unsigned short* __restrict__ Km,
    const unsigned short* __restrict__ Vt,
    unsigned short* __restrict__ O) {
  __shared__ unsigned short Ps[4][16][32];  // per-wave P staging (16 q x 32 k)

  const int t = threadIdx.x;
  const int wv = t >> 5, ln = t & 31;
  const int hl = ln >> 4, mrow = ln & 15, ncol = ln & 15;
  const int qtiles = S_ / 64;
  const int bh = blockIdx.x / qtiles;
  const int qt = blockIdx.x % qtiles;
  const int b = bh / H_, h = bh % H_;
  const int q0 = qt * 64 + wv * 16;
  const float scale = 0.125f;  // 1/sqrt(DH)

  // Q fragments for K-steps d=0..31 and d=32..63 (A layout, dword gathers)
  BF16Frag qa0, qa1;
  {
    const size_t rowbase = ((size_t)(b * S_ + q0 + mrow)) * D_ + h * DH_;
#pragma unroll
    for (int v = 0; v < 8; ++v) {
      int k = a_kbase(v, hl);
      qa0.u[v] = *(const unsigned int*)&Q[rowbase + k];
      qa1.u[v] = *(const unsigned int*)&Q[rowbase + 32 + k];
    }
  }

  v8f o0 = {}, o1 = {}, o2 = {}, o3 = {};
  float mrun[8], lrun[8];
#pragma unroll
  for (int r = 0; r < 8; ++r) { mrun[r] = -1e30f; lrun[r] = 0.f; }

  const unsigned short* Vtb = Vt + (size_t)bh * DH_ * S_;

  for (int kc = 0; kc < S_; kc += 32) {
    // Score B fragments: B = K^T (dh x key); lane = key column, half = dh half.
    // Element (dh = d0 + hl*16 + 2v+j, key = kc + 16*jsub + ncol): dh-pairs
    // are contiguous in K's natural [s][d] layout -> dword gathers (the
    // compiler merges the 8 consecutive dwords into global_load_b128).
    BF16Frag kb00, kb01, kb10, kb11;
    {
      const unsigned short* Kb =
          Km + ((size_t)(b * S_ + kc)) * D_ + h * DH_ + (hl << 4);
#pragma unroll
      for (int v = 0; v < 8; ++v) {
        int kk = v << 1;
        kb00.u[v] = *(const unsigned int*)&Kb[(size_t)ncol * D_ + kk];
        kb01.u[v] = *(const unsigned int*)&Kb[(size_t)(16 + ncol) * D_ + kk];
        kb10.u[v] = *(const unsigned int*)&Kb[(size_t)ncol * D_ + 32 + kk];
        kb11.u[v] = *(const unsigned int*)&Kb[(size_t)(16 + ncol) * D_ + 32 + kk];
      }
    }

    v8f z = {};
    v8f s0 = __builtin_amdgcn_wmma_f32_16x16x32_bf16(
        false, qa0.v, false, kb00.v, (short)0, z, false, false);
    s0 = __builtin_amdgcn_wmma_f32_16x16x32_bf16(
        false, qa1.v, false, kb10.v, (short)0, s0, false, false);
    v8f s1 = __builtin_amdgcn_wmma_f32_16x16x32_bf16(
        false, qa0.v, false, kb01.v, (short)0, z, false, false);
    s1 = __builtin_amdgcn_wmma_f32_16x16x32_bf16(
        false, qa1.v, false, kb11.v, (short)0, s1, false, false);

    // Online softmax per row (row = VGPR slot r + hl*8; cols across 16 lanes)
#pragma unroll
    for (int r = 0; r < 8; ++r) {
      float a0 = s0[r] * scale, a1 = s1[r] * scale;
      float mx = redmax16(fmaxf(a0, a1));
      float mn = fmaxf(mrun[r], mx);
      float alpha = __expf(mrun[r] - mn);
      float p0 = __expf(a0 - mn);
      float p1 = __expf(a1 - mn);
      lrun[r] = lrun[r] * alpha + redsum16(p0 + p1);
      mrun[r] = mn;
      o0[r] *= alpha; o1[r] *= alpha; o2[r] *= alpha; o3[r] *= alpha;
      int Mr = r + hl * 8;
      Ps[wv][Mr][ncol]      = f2bf(p0);
      Ps[wv][Mr][ncol + 16] = f2bf(p1);
    }
    __syncthreads();  // order cross-lane LDS stores before A-layout reads

    // P as A fragment (16x32) from LDS
    BF16Frag pa;
#pragma unroll
    for (int v = 0; v < 8; ++v)
      pa.u[v] = *(const unsigned int*)&Ps[wv][mrow][a_kbase(v, hl)];

    // V B fragments: B = V chunk (key x dh); lane = dh column, half = key half.
    // From Vt[b][h][dh][s]: key-pairs contiguous along s -> dword gathers.
    BF16Frag vb[4];
    {
#pragma unroll
      for (int j = 0; j < 4; ++j) {
#pragma unroll
        for (int v = 0; v < 8; ++v)
          vb[j].u[v] = *(const unsigned int*)
              &Vtb[(size_t)(j * 16 + ncol) * S_ + kc + (hl << 4) + (v << 1)];
      }
    }
    o0 = __builtin_amdgcn_wmma_f32_16x16x32_bf16(
        false, pa.v, false, vb[0].v, (short)0, o0, false, false);
    o1 = __builtin_amdgcn_wmma_f32_16x16x32_bf16(
        false, pa.v, false, vb[1].v, (short)0, o1, false, false);
    o2 = __builtin_amdgcn_wmma_f32_16x16x32_bf16(
        false, pa.v, false, vb[2].v, (short)0, o2, false, false);
    o3 = __builtin_amdgcn_wmma_f32_16x16x32_bf16(
        false, pa.v, false, vb[3].v, (short)0, o3, false, false);
    __syncthreads();  // protect Ps against next iteration's stores
  }

  // Finalize: divide by row sum, store concat-head layout O[b][s][h*DH+dh]
#pragma unroll
  for (int r = 0; r < 8; ++r) {
    float inv = 1.0f / lrun[r];
    int qrow = q0 + r + hl * 8;
    size_t obase = ((size_t)(b * S_ + qrow)) * D_ + h * DH_ + ncol;
    O[obase]      = f2bf(o0[r] * inv);
    O[obase + 16] = f2bf(o1[r] * inv);
    O[obase + 32] = f2bf(o2[r] * inv);
    O[obase + 48] = f2bf(o3[r] * inv);
  }
}

// ---------------------------------------------------------------------------
extern "C" void kernel_launch(void* const* d_in, const int* in_sizes, int n_in,
                              void* d_out, int out_size, void* d_ws,
                              size_t ws_size, hipStream_t stream) {
  (void)in_sizes; (void)n_in; (void)out_size; (void)ws_size;
  const float* x  = (const float*)d_in[0];
  const float* Wq = (const float*)d_in[1];
  const float* bq = (const float*)d_in[2];
  const float* Wk = (const float*)d_in[3];
  const float* bk = (const float*)d_in[4];
  const float* Wv = (const float*)d_in[5];
  const float* bv = (const float*)d_in[6];
  const float* Wo = (const float*)d_in[7];
  const float* bo = (const float*)d_in[8];
  float* out = (float*)d_out;

  char* ws = (char*)d_ws;
  const size_t MB = 1024 * 1024;
  unsigned short* xbf = (unsigned short*)(ws + 0);        // 16 MB
  unsigned short* wqb = (unsigned short*)(ws + 16 * MB);  // 2 MB
  unsigned short* wkb = (unsigned short*)(ws + 18 * MB);  // 2 MB
  unsigned short* wvb = (unsigned short*)(ws + 20 * MB);  // 2 MB
  unsigned short* wob = (unsigned short*)(ws + 22 * MB);  // 2 MB
  unsigned short* qbf = (unsigned short*)(ws + 24 * MB);  // 16 MB
  unsigned short* kbf = (unsigned short*)(ws + 40 * MB);  // 16 MB
  unsigned short* vbf = (unsigned short*)(ws + 56 * MB);  // 16 MB
  unsigned short* vtb = (unsigned short*)(ws + 72 * MB);  // 16 MB (total 88 MB)
  unsigned short* obf = xbf;  // x is dead once attention starts -> reuse

  const int nx = B_ * S_ * D_;  // 8388608
  const int nw = D_ * D_;       // 1048576

  cvt_f32_bf16<<<nx / 256, 256, 0, stream>>>(x,  xbf, nx);
  cvt_f32_bf16<<<nw / 256, 256, 0, stream>>>(Wq, wqb, nw);
  cvt_f32_bf16<<<nw / 256, 256, 0, stream>>>(Wk, wkb, nw);
  cvt_f32_bf16<<<nw / 256, 256, 0, stream>>>(Wv, wvb, nw);
  cvt_f32_bf16<<<nw / 256, 256, 0, stream>>>(Wo, wob, nw);

  dim3 g(B_ * S_ / 128, D_ / 64);  // 64 x 16 blocks
  gemm_bf16<true><<<g, 256, 0, stream>>>(xbf, wqb, bq, nullptr, qbf,
                                         B_ * S_, D_, D_);
  gemm_bf16<true><<<g, 256, 0, stream>>>(xbf, wkb, bk, nullptr, kbf,
                                         B_ * S_, D_, D_);
  gemm_bf16<true><<<g, 256, 0, stream>>>(xbf, wvb, bv, nullptr, vbf,
                                         B_ * S_, D_, D_);

  transpose_v<<<nx / 256, 256, 0, stream>>>(vbf, vtb);

  flash_attn<<<B_ * H_ * (S_ / 64), 128, 0, stream>>>(qbf, kbf, vtb, obf);

  gemm_bf16<false><<<g, 256, 0, stream>>>(obf, wob, bo, out, nullptr,
                                          B_ * S_, D_, D_);
}
